// MoE_10333691314728
// MI455X (gfx1250) — compile-verified, matrix operand order checked by reading, not run on hardware
//
#include <hip/hip_runtime.h>
#include <hip/hip_bf16.h>

#define DIMX   2048
#define INTERX 1408
#define T_TOK  4096
#define NE     15    // routed experts; index NE == shared expert
#define MT     32    // tokens per FFN block tile

typedef __attribute__((ext_vector_type(16))) __bf16 v16bf;
typedef __attribute__((ext_vector_type(8)))  float  v8f;

union Frag16 { v16bf v; uint4 q[2]; };

// ---- fp32 -> packed bf16x2 -------------------------------------------------
#if __has_builtin(__builtin_amdgcn_cvt_pk_bf16_f32)
static __device__ __forceinline__ unsigned pk2(float lo, float hi) {
    auto v = __builtin_amdgcn_cvt_pk_bf16_f32(lo, hi);
    return __builtin_bit_cast(unsigned, v);
}
#else
// round-to-nearest (+0x8000) then single v_perm_b32 pack of the two high halves
static __device__ __forceinline__ unsigned pk2(float lo, float hi) {
    return __builtin_amdgcn_perm(__float_as_uint(hi) + 0x8000u,
                                 __float_as_uint(lo) + 0x8000u, 0x07060302u);
}
#endif

static __device__ __forceinline__ unsigned short bf1(float f) {
    return (unsigned short)((__float_as_uint(f) + 0x8000u) >> 16);
}

// load 8 consecutive fp32, return 8 packed bf16 (16 bytes)
static __device__ __forceinline__ uint4 cvt8(const float* __restrict__ p) {
    float4 a = *(const float4*)p;
    float4 b = *(const float4*)(p + 4);
    uint4 r;
    r.x = pk2(a.x, a.y); r.y = pk2(a.z, a.w);
    r.z = pk2(b.x, b.y); r.w = pk2(b.z, b.w);
    return r;
}

static __device__ __forceinline__ float silu_mul(float g, float u) {
    return (g / (1.0f + __expf(-g))) * u;
}

// ---------------- elementwise fp32 -> bf16 (grid-stride) ----------------
__global__ __launch_bounds__(256) void cvt_bf16_kernel(const float* __restrict__ src,
                                                       unsigned* __restrict__ dst, int npairs) {
    const float2* s = (const float2*)src;
    for (int i = blockIdx.x * blockDim.x + threadIdx.x; i < npairs;
         i += gridDim.x * blockDim.x) {
        float2 v = s[i];
        dst[i] = pk2(v.x, v.y);
    }
}

// ---------------- gate: softmax + top-2, build per-expert token lists ----------------
__global__ __launch_bounds__(128) void moe_gate_kernel(
    const float* __restrict__ x, const float* __restrict__ gw,
    int* __restrict__ cnt, int* __restrict__ tokIdx, float* __restrict__ tokW) {
    const int lane = threadIdx.x & 31;
    const int t = blockIdx.x * 4 + (threadIdx.x >> 5);   // one wave per token
    const float* xr = x + (size_t)t * DIMX;

    float logit[NE];
    for (int e = 0; e < NE; ++e) {
        const float* g = gw + (size_t)e * DIMX;
        float s = 0.f;
        for (int k = lane; k < DIMX; k += 32) s += xr[k] * g[k];
        for (int off = 16; off > 0; off >>= 1) s += __shfl_xor(s, off, 32);
        logit[e] = s;                                    // all lanes hold the sum
    }
    float mx = logit[0];
    for (int e = 1; e < NE; ++e) mx = fmaxf(mx, logit[e]);
    float p[NE], denom = 0.f;
    for (int e = 0; e < NE; ++e) { p[e] = __expf(logit[e] - mx); denom += p[e]; }
    const float inv = 1.f / denom;
    int i1 = 0, i2 = 0; float p1 = -1.f, p2 = -1.f;
    for (int e = 0; e < NE; ++e) {
        float pe = p[e] * inv;
        if (pe > p1)      { p2 = p1; i2 = i1; p1 = pe; i1 = e; }
        else if (pe > p2) { p2 = pe; i2 = e; }
    }
    if (lane == 0) {
        int q1 = atomicAdd(&cnt[i1], 1);
        tokIdx[i1 * T_TOK + q1] = t; tokW[i1 * T_TOK + q1] = p1;
        int q2 = atomicAdd(&cnt[i2], 1);
        tokIdx[i2 * T_TOK + q2] = t; tokW[i2 * T_TOK + q2] = p2;
    }
}

// B-fragment load: BPRE -> bf16 straight b128 loads; else fp32 + perm-pack
template <bool BPRE>
static __device__ __forceinline__ void loadB(Frag16& b, const void* __restrict__ row,
                                             int k, int kbase) {
    if constexpr (BPRE) {
        const unsigned short* p = (const unsigned short*)row;
        b.q[0] = *(const uint4*)(p + k + kbase);
        b.q[1] = *(const uint4*)(p + k + kbase + 16);
    } else {
        const float* p = (const float*)row;
        b.q[0] = cvt8(p + k + kbase);
        b.q[1] = cvt8(p + k + kbase + 16);
    }
}

// ---------------- fused SwiGLU expert FFN: 32-token tile per block ----------------
template <bool BPRE>
__global__ __launch_bounds__(256) void moe_ffn_kernel(
    const unsigned short* __restrict__ xb,
    const void* __restrict__ w1, const void* __restrict__ w3, const void* __restrict__ w2,
    const void* __restrict__ sw1, const void* __restrict__ sw3, const void* __restrict__ sw2,
    const int* __restrict__ cnt, const int* __restrict__ tokIdx, const float* __restrict__ tokW,
    float* __restrict__ out) {
    __shared__ unsigned short hS[MT * INTERX];   // 88 KB bf16 H tile (LDS: 320 KB/WGP)
    __shared__ int   tokS[MT];
    __shared__ float wgtS[MT];

    const int e = blockIdx.y;                    // 0..14 routed, 15 shared
    const int tile = blockIdx.x;
    const int rows = (e == NE) ? T_TOK : cnt[e];
    if (tile * MT >= rows) return;               // uniform exit, EXEC stays full

    const int tid = threadIdx.x;
    if (tid < MT) {
        int r = tile * MT + tid;
        int tok; float wg;
        if (e == NE)       { tok = r; wg = 1.0f; }
        else if (r < rows) { tok = tokIdx[e * T_TOK + r]; wg = tokW[e * T_TOK + r]; }
        else               { tok = 0; wg = 0.0f; }       // pad rows: weight 0
        tokS[tid] = tok; wgtS[tid] = wg;
    }
    __syncthreads();

    const int ebytes = BPRE ? 2 : 4;             // element size of W storage
    const char* W1 = (const char*)((e == NE) ? sw1 : w1) +
                     (e == NE ? 0 : (size_t)e * INTERX * DIMX * ebytes);
    const char* W3 = (const char*)((e == NE) ? sw3 : w3) +
                     (e == NE ? 0 : (size_t)e * INTERX * DIMX * ebytes);
    const char* W2 = (const char*)((e == NE) ? sw2 : w2) +
                     (e == NE ? 0 : (size_t)e * DIMX * INTERX * ebytes);

    const int lane  = tid & 31;
    const int wv    = tid >> 5;        // wave 0..7
    const int half  = lane >> 4;       // K-half select per 16-bit A/B VGPR layout
    const int mrow  = lane & 15;       // A-row / B-col inside 16x16 tile
    const int kbase = half * 8;

    const unsigned short* arow0 = xb + (size_t)tokS[mrow] * DIMX;        // rows 0..15
    const unsigned short* arow1 = xb + (size_t)tokS[16 + mrow] * DIMX;   // rows 16..31

    // ---- pass 1: H[32 x 1408] = silu(X*W1^T) * (X*W3^T); each wave 11 column tiles
    for (int t = 0; t < 11; ++t) {
        const int n0 = (wv * 11 + t) * 16;
        const void* b1row = W1 + (size_t)(n0 + mrow) * DIMX * ebytes;
        const void* b3row = W3 + (size_t)(n0 + mrow) * DIMX * ebytes;
        v8f a1lo = {}, a1hi = {}, a3lo = {}, a3hi = {};
        for (int k = 0; k < DIMX; k += 32) {
            Frag16 a0, a1, b1, b3;
            a0.q[0] = *(const uint4*)(arow0 + k + kbase);
            a0.q[1] = *(const uint4*)(arow0 + k + kbase + 16);
            a1.q[0] = *(const uint4*)(arow1 + k + kbase);
            a1.q[1] = *(const uint4*)(arow1 + k + kbase + 16);
            loadB<BPRE>(b1, b1row, k, kbase);
            loadB<BPRE>(b3, b3row, k, kbase);
            a1lo = __builtin_amdgcn_wmma_f32_16x16x32_bf16(false, a0.v, false, b1.v, (short)0, a1lo, false, false);
            a3lo = __builtin_amdgcn_wmma_f32_16x16x32_bf16(false, a0.v, false, b3.v, (short)0, a3lo, false, false);
            a1hi = __builtin_amdgcn_wmma_f32_16x16x32_bf16(false, a1.v, false, b1.v, (short)0, a1hi, false, false);
            a3hi = __builtin_amdgcn_wmma_f32_16x16x32_bf16(false, a1.v, false, b3.v, (short)0, a3hi, false, false);
        }
        const int n = n0 + mrow;       // D layout: lane -> col n, VGPR i -> row i+8*half
        #pragma unroll
        for (int i = 0; i < 8; ++i) {
            const int m = i + 8 * half;
            hS[(size_t)m * INTERX + n]        = bf1(silu_mul(a1lo[i], a3lo[i]));
            hS[(size_t)(16 + m) * INTERX + n] = bf1(silu_mul(a1hi[i], a3hi[i]));
        }
    }
    __syncthreads();

    // ---- pass 2: Y[32 x 2048] = H * W2^T, scale by gate prob, scatter-add
    const unsigned short* hrow0 = hS + (size_t)mrow * INTERX;
    const unsigned short* hrow1 = hS + (size_t)(16 + mrow) * INTERX;
    for (int t = 0; t < 16; ++t) {
        const int n0 = (wv * 16 + t) * 16;
        const void* b2row = W2 + (size_t)(n0 + mrow) * INTERX * ebytes;
        v8f acclo = {}, acchi = {};
        for (int k = 0; k < INTERX; k += 32) {
            Frag16 a0, a1, b;
            a0.q[0] = *(const uint4*)(hrow0 + k + kbase);
            a0.q[1] = *(const uint4*)(hrow0 + k + kbase + 16);
            a1.q[0] = *(const uint4*)(hrow1 + k + kbase);
            a1.q[1] = *(const uint4*)(hrow1 + k + kbase + 16);
            loadB<BPRE>(b, b2row, k, kbase);
            acclo = __builtin_amdgcn_wmma_f32_16x16x32_bf16(false, a0.v, false, b.v, (short)0, acclo, false, false);
            acchi = __builtin_amdgcn_wmma_f32_16x16x32_bf16(false, a1.v, false, b.v, (short)0, acchi, false, false);
        }
        const int n = n0 + mrow;
        #pragma unroll
        for (int i = 0; i < 8; ++i) {
            const int m = i + 8 * half;
            atomicAdd(&out[(size_t)tokS[m] * DIMX + n],      acclo[i] * wgtS[m]);
            atomicAdd(&out[(size_t)tokS[16 + m] * DIMX + n], acchi[i] * wgtS[16 + m]);
        }
    }
}

extern "C" void kernel_launch(void* const* d_in, const int* in_sizes, int n_in,
                              void* d_out, int out_size, void* d_ws, size_t ws_size,
                              hipStream_t stream) {
    const float* x      = (const float*)d_in[0];
    const float* gate_w = (const float*)d_in[1];
    const float* w1     = (const float*)d_in[2];
    const float* w3     = (const float*)d_in[3];
    const float* w2     = (const float*)d_in[4];
    const float* sw1    = (const float*)d_in[5];
    const float* sw3    = (const float*)d_in[6];
    const float* sw2    = (const float*)d_in[7];
    float* out = (float*)d_out;

    // ---- workspace layout ----
    const size_t SZ_XB   = (size_t)T_TOK * DIMX * 2;          // 16.78 MB
    const size_t SZ_CNT  = 64;
    const size_t SZ_IDX  = (size_t)NE * T_TOK * 4;
    const size_t SZ_W    = (size_t)NE * INTERX * DIMX * 2;    // 86.5 MB each
    const size_t SZ_SW   = (size_t)INTERX * DIMX * 2;         // 5.77 MB each
    const size_t SMALL   = SZ_XB + SZ_CNT + 2 * SZ_IDX;
    const size_t FULL    = SMALL + 3 * SZ_W + 3 * SZ_SW;      // ~281 MiB

    char* ws = (char*)d_ws;
    unsigned short* xb = (unsigned short*)ws;
    int*   cnt    = (int*)(ws + SZ_XB);
    int*   tokIdx = (int*)(ws + SZ_XB + SZ_CNT);
    float* tokW   = (float*)(ws + SZ_XB + SZ_CNT + SZ_IDX);

    const bool pre = (ws_size >= FULL);

    hipMemsetAsync(out, 0, (size_t)out_size * sizeof(float), stream);
    hipMemsetAsync(cnt, 0, 64, stream);

    // x -> bf16 (always)
    cvt_bf16_kernel<<<4096, 256, 0, stream>>>(x, (unsigned*)xb, T_TOK * DIMX / 2);
    moe_gate_kernel<<<T_TOK / 4, 128, 0, stream>>>(x, gate_w, cnt, tokIdx, tokW);

    if (pre) {
        char* p = ws + SMALL;
        unsigned short *w1b = (unsigned short*)p;            p += SZ_W;
        unsigned short *w3b = (unsigned short*)p;            p += SZ_W;
        unsigned short *w2b = (unsigned short*)p;            p += SZ_W;
        unsigned short *s1b = (unsigned short*)p;            p += SZ_SW;
        unsigned short *s3b = (unsigned short*)p;            p += SZ_SW;
        unsigned short *s2b = (unsigned short*)p;
        const int nwp = NE * INTERX * DIMX / 2;
        const int nsp = INTERX * DIMX / 2;
        cvt_bf16_kernel<<<8192, 256, 0, stream>>>(w1,  (unsigned*)w1b, nwp);
        cvt_bf16_kernel<<<8192, 256, 0, stream>>>(w3,  (unsigned*)w3b, nwp);
        cvt_bf16_kernel<<<8192, 256, 0, stream>>>(w2,  (unsigned*)w2b, nwp);
        cvt_bf16_kernel<<<2048, 256, 0, stream>>>(sw1, (unsigned*)s1b, nsp);
        cvt_bf16_kernel<<<2048, 256, 0, stream>>>(sw3, (unsigned*)s3b, nsp);
        cvt_bf16_kernel<<<2048, 256, 0, stream>>>(sw2, (unsigned*)s2b, nsp);
        moe_ffn_kernel<true><<<dim3(T_TOK / MT, NE + 1), 256, 0, stream>>>(
            xb, w1b, w3b, w2b, s1b, s3b, s2b, cnt, tokIdx, tokW, out);
    } else {
        moe_ffn_kernel<false><<<dim3(T_TOK / MT, NE + 1), 256, 0, stream>>>(
            xb, w1, w3, w2, sw1, sw3, sw2, cnt, tokIdx, tokW, out);
    }
}
